// MultiHeadAttention_40389872451632
// MI455X (gfx1250) — matrix operand order repaired and checked by reading, not compiled, not run on hardware
//
#include <hip/hip_runtime.h>
#include <hip/hip_bf16.h>
#include <math.h>

// MI455X / gfx1250, wave32. Flash-attention MHA forward, all GEMMs on
// v_wmma_f32_16x16x32_bf16.
// Workspace (bf16): Xq,Xk,Xv [B,S,D] (24MB) + Wq,Wk,Wv,Wo (8MB) +
// Q,K [B,H,S,DK] (16MB) + V^T [B,H,DK,S] (8MB) + ctx [B,S,D] (8MB) = 64MB.
// Whole problem (117MB IO + ws) is L2-resident (192MB); compute-bound on
// WMMA issue, so maximize wmma per loaded byte via 32x32 register tiles.

#define B_  2
#define S_  2048
#define D_  1024
#define H_  16
#define DK_ 64
#define BS_ (B_ * S_)

typedef __bf16 bf16_t;
typedef __attribute__((ext_vector_type(16))) __bf16 v16bf;
typedef __attribute__((ext_vector_type(8)))  __bf16 v8bf;
typedef __attribute__((ext_vector_type(8)))  float  v8f;

static __device__ __forceinline__ v8f wmma_bf16(v16bf a, v16bf b, v8f c) {
  // (neg_a, A, neg_b, B, c_mod, C, reuse_a, reuse_b)
  return __builtin_amdgcn_wmma_f32_16x16x32_bf16(false, a, false, b,
                                                 (short)0, c, false, false);
}

// A-operand (16x32 bf16): lane holds row, K chunks [p..p+7] and [p+16..p+23]
// (caller bakes in hi*8).
static __device__ __forceinline__ v16bf pack_a_bf(const bf16_t* p) {
  v8bf x0 = *(const v8bf*)(p);
  v8bf x1 = *(const v8bf*)(p + 16);
  v16bf a;
#pragma unroll
  for (int i = 0; i < 8; ++i) { a[i] = x0[i]; a[8 + i] = x1[i]; }
  return a;
}

// B-operand (32x16 bf16): lane = column, K chunks [p..p+7] and [p+16..p+23]
// (caller bakes in hi*8).
static __device__ __forceinline__ v16bf pack_b_bf(const bf16_t* p) {
  v8bf x0 = *(const v8bf*)(p);
  v8bf x1 = *(const v8bf*)(p + 16);
  v16bf b;
#pragma unroll
  for (int i = 0; i < 8; ++i) { b[i] = x0[i]; b[8 + i] = x1[i]; }
  return b;
}

// ---------------- Kernel 0: one-shot f32 -> bf16 conversion ---------------
__global__ __launch_bounds__(256) void cvt_bf16_kernel(
    const float* __restrict__ src, bf16_t* __restrict__ dst, int n) {
  const int i = (blockIdx.x * blockDim.x + threadIdx.x) * 8;
  if (i >= n) return;
  float4 x0 = *(const float4*)(src + i);
  float4 x1 = *(const float4*)(src + i + 4);
  v8bf o;
  o[0]=(bf16_t)x0.x; o[1]=(bf16_t)x0.y; o[2]=(bf16_t)x0.z; o[3]=(bf16_t)x0.w;
  o[4]=(bf16_t)x1.x; o[5]=(bf16_t)x1.y; o[6]=(bf16_t)x1.z; o[7]=(bf16_t)x1.w;
  *(v8bf*)(dst + i) = o;
}

// ---------------- Kernel 1: fused QKV projection (32x32 tiles) ------------
// One wave per 32x32 tile of Y = X @ W^T + b. Q,K -> [B,H,S,DK]; V -> V^T
// [B,H,DK,S] so attention's P@V B-operand loads are contiguous.
__global__ __launch_bounds__(256) void qkv_proj_kernel(
    const bf16_t* __restrict__ Xq, const bf16_t* __restrict__ Xk,
    const bf16_t* __restrict__ Xv,
    const bf16_t* __restrict__ Wqb, const float* __restrict__ bq,
    const bf16_t* __restrict__ Wkb, const float* __restrict__ bk,
    const bf16_t* __restrict__ Wvb, const float* __restrict__ bv,
    bf16_t* __restrict__ Qb, bf16_t* __restrict__ Kb,
    bf16_t* __restrict__ Vt) {
  const int lane = threadIdx.x & 31;
  const int wave = (blockIdx.x * blockDim.x + threadIdx.x) >> 5;
  const int nTiles = D_ / 32;                   // 32
  const int tilesPerMat = (BS_ / 32) * nTiles;  // 4096
  const int m  = wave / tilesPerMat;            // 0=Q,1=K,2=V
  const int t  = wave - m * tilesPerMat;
  const int nT = t % nTiles;
  const int mT = t / nTiles;

  const bf16_t* X    = (m == 0) ? Xq  : (m == 1) ? Xk  : Xv;
  const bf16_t* W    = (m == 0) ? Wqb : (m == 1) ? Wkb : Wvb;
  const float*  bias = (m == 0) ? bq  : (m == 1) ? bk  : bv;
  bf16_t*       out  = (m == 0) ? Qb  : Kb;  // m==2 handled via Vt below

  const int hi = lane >> 4, l15 = lane & 15;
  const bf16_t* Arow0 = X + (size_t)(mT * 32 + l15) * D_;
  const bf16_t* Arow1 = Arow0 + (size_t)16 * D_;
  const bf16_t* Brow0 = W + (size_t)(nT * 32 + l15) * D_;
  const bf16_t* Brow1 = Brow0 + (size_t)16 * D_;

  v8f zero = {};
  v8f acc[2][2] = {{zero, zero}, {zero, zero}};
  for (int k0 = 0; k0 < D_; k0 += 32) {
    v16bf a0 = pack_a_bf(Arow0 + k0 + hi * 8);
    v16bf a1 = pack_a_bf(Arow1 + k0 + hi * 8);
    v16bf b0 = pack_b_bf(Brow0 + k0 + hi * 8);
    v16bf b1 = pack_b_bf(Brow1 + k0 + hi * 8);
    acc[0][0] = wmma_bf16(a0, b0, acc[0][0]);
    acc[0][1] = wmma_bf16(a0, b1, acc[0][1]);
    acc[1][0] = wmma_bf16(a1, b0, acc[1][0]);
    acc[1][1] = wmma_bf16(a1, b1, acc[1][1]);
  }

#pragma unroll
  for (int ai = 0; ai < 2; ++ai) {
#pragma unroll
    for (int bi = 0; bi < 2; ++bi) {
      const int n = nT * 32 + bi * 16 + l15;   // output column
      const float bb = bias[n];
      const int hcol = n >> 6, dcol = n & 63;
#pragma unroll
      for (int j = 0; j < 8; ++j) {            // C layout: VGPR j -> rows j/j+8
        const int r = mT * 32 + ai * 16 + j + hi * 8;  // row in [B*S]
        const int bidx = r >> 11, srow = r & (S_ - 1);
        const bf16_t val = (bf16_t)(acc[ai][bi][j] + bb);
        if (m == 2)
          Vt[(((size_t)bidx * H_ + hcol) * DK_ + dcol) * S_ + srow] = val;
        else
          out[(((size_t)bidx * H_ + hcol) * S_ + srow) * DK_ + dcol] = val;
      }
    }
  }
}

// ---------------- Kernel 2: flash attention -------------------------------
// One wave per (b, h, 16-row q-tile). Streams keys 32 at a time with online
// softmax; P transposed C-layout -> A-layout through per-wave LDS; V^T gives
// contiguous B-operand loads for P@V.
__global__ __launch_bounds__(256) void attention_kernel(
    const bf16_t* __restrict__ Qb, const bf16_t* __restrict__ Kb,
    const bf16_t* __restrict__ Vt, const int* __restrict__ mask,
    bf16_t* __restrict__ Cb) {
  __shared__ bf16_t Pall[8][16 * 40];  // 16x32 P tile per wave, 16B-aligned rows
  const int lane  = threadIdx.x & 31;
  const int wslot = threadIdx.x >> 5;
  const int wave  = (blockIdx.x * blockDim.x + threadIdx.x) >> 5;
  const int qt = wave & 127;           // S/16 = 128 q-tiles
  const int bh = wave >> 7;
  const int h = bh & (H_ - 1), b = bh >> 4;
  const int hi = lane >> 4, l15 = lane & 15;

  const bf16_t* Qbase  = Qb + (size_t)(b * H_ + h) * S_ * DK_;
  const bf16_t* Kbase  = Kb + (size_t)(b * H_ + h) * S_ * DK_;
  const bf16_t* Vtbase = Vt + (size_t)(b * H_ + h) * DK_ * S_;
  const int* Mbase = mask + (size_t)b * S_ * S_;

  const int qrow = qt * 16 + l15;
  // Q tile 16x64 as two A-operands (contraction d 0..31 and 32..63)
  v16bf aq0 = pack_a_bf(Qbase + (size_t)qrow * DK_ + hi * 8);
  v16bf aq1 = pack_a_bf(Qbase + (size_t)qrow * DK_ + 32 + hi * 8);

  float mrow[8], lrow[8];
#pragma unroll
  for (int j = 0; j < 8; ++j) { mrow[j] = -1e30f; lrow[j] = 0.f; }
  v8f zero = {};
  v8f acc[4] = {zero, zero, zero, zero};  // ctx 16x64 = 4 N-tiles
  bf16_t* P = &Pall[wslot][0];

  for (int j0 = 0; j0 < S_; j0 += 32) {
    float s[2][8];
#pragma unroll
    for (int t = 0; t < 2; ++t) {
      const int kb16 = j0 + t * 16;
      const bf16_t* kp = Kbase + (size_t)(kb16 + l15) * DK_;  // lane = key col
      v16bf bkA = pack_b_bf(kp + hi * 8);       // d 0..31
      v16bf bkB = pack_b_bf(kp + 32 + hi * 8);  // d 32..63
      v8f sc = {};
      sc = wmma_bf16(aq0, bkA, sc);
      sc = wmma_bf16(aq1, bkB, sc);
      const int kc = kb16 + l15;
#pragma unroll
      for (int j = 0; j < 8; ++j) {
        const int qr = qt * 16 + j + hi * 8;
        const int mv = Mbase[(size_t)qr * S_ + kc];
        s[t][j] = (mv == 0) ? -1e9f : sc[j] * 0.125f;  // 1/sqrt(64)
      }
    }
    // online softmax: row reductions across each 16-lane group
    float mnew[8], sf[8];
#pragma unroll
    for (int j = 0; j < 8; ++j) {
      float r = fmaxf(s[0][j], s[1][j]);
      r = fmaxf(r, __shfl_xor(r, 1, 32));
      r = fmaxf(r, __shfl_xor(r, 2, 32));
      r = fmaxf(r, __shfl_xor(r, 4, 32));
      r = fmaxf(r, __shfl_xor(r, 8, 32));
      mnew[j] = fmaxf(mrow[j], r);
      sf[j] = __expf(mrow[j] - mnew[j]);
      mrow[j] = mnew[j];
    }
#pragma unroll
    for (int t = 0; t < 2; ++t)
#pragma unroll
      for (int j = 0; j < 8; ++j) {
        float p = __expf(s[t][j] - mnew[j]);
        s[t][j] = p;
        P[(j + hi * 8) * 40 + t * 16 + l15] = (bf16_t)p;  // C-layout -> LDS
      }
#pragma unroll
    for (int j = 0; j < 8; ++j) {
      float r = s[0][j] + s[1][j];
      r += __shfl_xor(r, 1, 32);
      r += __shfl_xor(r, 2, 32);
      r += __shfl_xor(r, 4, 32);
      r += __shfl_xor(r, 8, 32);
      lrow[j] = lrow[j] * sf[j] + r;
    }
#pragma unroll
    for (int n = 0; n < 4; ++n)
#pragma unroll
      for (int j = 0; j < 8; ++j) acc[n][j] *= sf[j];

    // same-wave LDS ordering: wait for P stores before transposed reads
    asm volatile("s_wait_dscnt 0" ::: "memory");
    v8bf p0 = *(const v8bf*)(P + l15 * 40 + hi * 8);
    v8bf p1 = *(const v8bf*)(P + l15 * 40 + 16 + hi * 8);
    v16bf pa;
#pragma unroll
    for (int i = 0; i < 8; ++i) { pa[i] = p0[i]; pa[8 + i] = p1[i]; }

    // ctx += P(16x32) @ V(32x64); V^T row = d, keys contiguous
#pragma unroll
    for (int n = 0; n < 4; ++n) {
      const bf16_t* vrow = Vtbase + (size_t)(n * 16 + l15) * S_ + j0 + hi * 8;
      v16bf bvt = pack_b_bf(vrow);
      acc[n] = wmma_bf16(pa, bvt, acc[n]);
    }
    asm volatile("s_wait_dscnt 0" ::: "memory");  // reads done before next stores
  }

  // epilogue: ctx/l -> merged-head bf16 [B,S,D]
#pragma unroll
  for (int j = 0; j < 8; ++j) lrow[j] = 1.0f / lrow[j];
#pragma unroll
  for (int n = 0; n < 4; ++n) {
    const int col = h * DK_ + n * 16 + l15;
#pragma unroll
    for (int j = 0; j < 8; ++j) {
      const int qr = qt * 16 + j + hi * 8;
      Cb[((size_t)b * S_ + qr) * D_ + col] = (bf16_t)(acc[n][j] * lrow[j]);
    }
  }
}

// ---------------- Kernel 3: output projection (32x32 tiles) ---------------
__global__ __launch_bounds__(256) void out_proj_kernel(
    const bf16_t* __restrict__ Cb, const bf16_t* __restrict__ Wob,
    const float* __restrict__ bo, float* __restrict__ out) {
  const int lane = threadIdx.x & 31;
  const int wave = (blockIdx.x * blockDim.x + threadIdx.x) >> 5;
  const int nTiles = D_ / 32;
  const int nT = wave % nTiles;
  const int mT = wave / nTiles;
  const int hi = lane >> 4, l15 = lane & 15;
  const bf16_t* Arow0 = Cb + (size_t)(mT * 32 + l15) * D_;
  const bf16_t* Arow1 = Arow0 + (size_t)16 * D_;
  const bf16_t* Brow0 = Wob + (size_t)(nT * 32 + l15) * D_;
  const bf16_t* Brow1 = Brow0 + (size_t)16 * D_;

  v8f zero = {};
  v8f acc[2][2] = {{zero, zero}, {zero, zero}};
  for (int k0 = 0; k0 < D_; k0 += 32) {
    v16bf a0 = pack_a_bf(Arow0 + k0 + hi * 8);
    v16bf a1 = pack_a_bf(Arow1 + k0 + hi * 8);
    v16bf b0 = pack_b_bf(Brow0 + k0 + hi * 8);
    v16bf b1 = pack_b_bf(Brow1 + k0 + hi * 8);
    acc[0][0] = wmma_bf16(a0, b0, acc[0][0]);
    acc[0][1] = wmma_bf16(a0, b1, acc[0][1]);
    acc[1][0] = wmma_bf16(a1, b0, acc[1][0]);
    acc[1][1] = wmma_bf16(a1, b1, acc[1][1]);
  }
#pragma unroll
  for (int ai = 0; ai < 2; ++ai) {
#pragma unroll
    for (int bi = 0; bi < 2; ++bi) {
      const int n = nT * 32 + bi * 16 + l15;
      const float bb = bo[n];
#pragma unroll
      for (int j = 0; j < 8; ++j) {
        const int r = mT * 32 + ai * 16 + j + hi * 8;
        out[(size_t)r * D_ + n] = acc[ai][bi][j] + bb;
      }
    }
  }
}

extern "C" void kernel_launch(void* const* d_in, const int* in_sizes, int n_in,
                              void* d_out, int out_size, void* d_ws,
                              size_t ws_size, hipStream_t stream) {
  (void)in_sizes; (void)n_in; (void)out_size; (void)ws_size;
  const float* query = (const float*)d_in[0];
  const float* key_  = (const float*)d_in[1];
  const float* value = (const float*)d_in[2];
  const int*   mask  = (const int*)d_in[3];
  const float* Wq = (const float*)d_in[4];
  const float* bq = (const float*)d_in[5];
  const float* Wk = (const float*)d_in[6];
  const float* bk = (const float*)d_in[7];
  const float* Wv = (const float*)d_in[8];
  const float* bv = (const float*)d_in[9];
  const float* Wo = (const float*)d_in[10];
  const float* bo = (const float*)d_in[11];
  float* out = (float*)d_out;

  const size_t xE = (size_t)BS_ * D_;  // 4Mi elems
  const size_t wE = (size_t)D_ * D_;   // 1Mi elems
  bf16_t* p = (bf16_t*)d_ws;
  bf16_t* Xq = p;  p += xE;
  bf16_t* Xk = p;  p += xE;
  bf16_t* Xv = p;  p += xE;
  bf16_t* Wqb = p; p += wE;
  bf16_t* Wkb = p; p += wE;
  bf16_t* Wvb = p; p += wE;
  bf16_t* Wob = p; p += wE;
  bf16_t* Qb = p;  p += xE;
  bf16_t* Kb = p;  p += xE;
  bf16_t* Vt = p;  p += xE;   // [B,H,DK,S]
  bf16_t* Cb = p;             // total 64MB

  // 0) one-shot bf16 conversion (activations + weights)
  cvt_bf16_kernel<<<(int)(xE / 8 / 256), 256, 0, stream>>>(query, Xq, (int)xE);
  cvt_bf16_kernel<<<(int)(xE / 8 / 256), 256, 0, stream>>>(key_,  Xk, (int)xE);
  cvt_bf16_kernel<<<(int)(xE / 8 / 256), 256, 0, stream>>>(value, Xv, (int)xE);
  cvt_bf16_kernel<<<(int)(wE / 8 / 256), 256, 0, stream>>>(Wq, Wqb, (int)wE);
  cvt_bf16_kernel<<<(int)(wE / 8 / 256), 256, 0, stream>>>(Wk, Wkb, (int)wE);
  cvt_bf16_kernel<<<(int)(wE / 8 / 256), 256, 0, stream>>>(Wv, Wvb, (int)wE);
  cvt_bf16_kernel<<<(int)(wE / 8 / 256), 256, 0, stream>>>(Wo, Wob, (int)wE);

  // 1) QKV: 3 * 128 * 32 tiles = 12288 waves, 8 waves/block
  {
    const int waves = 3 * (BS_ / 32) * (D_ / 32);
    qkv_proj_kernel<<<waves / 8, 256, 0, stream>>>(
        Xq, Xk, Xv, Wqb, bq, Wkb, bk, Wvb, bv, Qb, Kb, Vt);
  }
  // 2) attention: B*H*(S/16) = 4096 waves
  {
    const int waves = B_ * H_ * (S_ / 16);
    attention_kernel<<<waves / 8, 256, 0, stream>>>(Qb, Kb, Vt, mask, Cb);
  }
  // 3) output projection: 128 * 32 = 4096 waves
  {
    const int waves = (BS_ / 32) * (D_ / 32);
    out_proj_kernel<<<waves / 8, 256, 0, stream>>>(Cb, Wob, bo, out);
  }
}
